// MoE_67242007986677
// MI455X (gfx1250) — compile-verified, matrix operand order checked by reading, not run on hardware
//
#include <hip/hip_runtime.h>

typedef unsigned short u16;
typedef __bf16 bf16_t;
typedef bf16_t v16bf __attribute__((ext_vector_type(16)));
typedef bf16_t v8bf  __attribute__((ext_vector_type(8)));
typedef float  v8f   __attribute__((ext_vector_type(8)));
typedef int    v4i   __attribute__((ext_vector_type(4)));

#define D_HID 1024
#define D_EXP 512
#define N_EXP 8
#define D_SH  1024
#define LPAD  40   // padded LDS row stride (ushorts) -> conflict-free b128 reads
#define KSTEP 32

// ---- CDNA5 async global->LDS path (guarded; falls back to uint4 staging) ----
#if defined(__has_builtin)
# if __has_builtin(__builtin_amdgcn_global_load_async_to_lds_b128) && \
     __has_builtin(__builtin_amdgcn_s_wait_asynccnt)
#  define USE_ASYNC_LDS 1
# endif
#endif
#ifndef USE_ASYNC_LDS
# define USE_ASYNC_LDS 0
#endif

#if USE_ASYNC_LDS
#define AS1 __attribute__((address_space(1)))
#define AS3 __attribute__((address_space(3)))
#endif

__device__ __forceinline__ void stage16(u16* dst, const u16* src) {
#if USE_ASYNC_LDS
  __builtin_amdgcn_global_load_async_to_lds_b128(
      (AS1 v4i*)(void*)src, (AS3 v4i*)(void*)dst, 0, 0);
#else
  *(uint4*)dst = *(const uint4*)src;
#endif
}

template <int N>
__device__ __forceinline__ void wait_stage() {
#if USE_ASYNC_LDS
  __builtin_amdgcn_s_wait_asynccnt(N);
#endif
}

__device__ __forceinline__ u16 f2bf(float f) {
  unsigned u = __float_as_uint(f);
  u += 0x7FFFu + ((u >> 16) & 1u);   // round-to-nearest-even
  return (u16)(u >> 16);
}

union FragBF { v16bf v; v8bf h[2]; };

__device__ __forceinline__ v8f wmma_bf16(v16bf a, v16bf b, v8f c) {
  return __builtin_amdgcn_wmma_f32_16x16x32_bf16(
      false, a, false, b, (short)0, c, false, false);
}

// ---------------- small utility kernels ----------------

__global__ void k_zero(int* p, int n) {
  int i = blockIdx.x * blockDim.x + threadIdx.x;
  if (i < n) p[i] = 0;
}

__global__ void k_f32_to_bf16(const float* __restrict__ in, u16* __restrict__ out, int n) {
  int i = blockIdx.x * blockDim.x + threadIdx.x;
  if (i < n) out[i] = f2bf(in[i]);
}

// in: [batch][R][C] fp32  ->  out: [batch][C][R] bf16
__global__ __launch_bounds__(256) void k_transpose_bf16(
    const float* __restrict__ in, u16* __restrict__ out, int R, int C) {
  __shared__ u16 tile[32][33];
  size_t bo = (size_t)blockIdx.z * R * C;
  int c0 = blockIdx.x * 32, r0 = blockIdx.y * 32;
  int tx = threadIdx.x & 31, ty = threadIdx.x >> 5;
#pragma unroll
  for (int i = ty; i < 32; i += 8)
    tile[i][tx] = f2bf(in[bo + (size_t)(r0 + i) * C + c0 + tx]);
  __syncthreads();
#pragma unroll
  for (int i = ty; i < 32; i += 8)
    out[bo + (size_t)(c0 + i) * R + r0 + tx] = tile[tx][i];
}

// ---------------- gating: softmax + top-2 + per-expert compaction ----------------
// One wave per token. Appends slot = 2*token+k and weight to expert i's list.
// Atomic order only affects tile placement, never values -> deterministic output.
__global__ __launch_bounds__(256) void k_gate(
    const float* __restrict__ x, const float* __restrict__ Wg,
    int* __restrict__ cnt, int* __restrict__ slotList, float* __restrict__ wList, int N) {
  int wid = threadIdx.x >> 5, lane = threadIdx.x & 31;
  int tok = blockIdx.x * 8 + wid;
  if (tok >= N) return;
  float l[8] = {0.f, 0.f, 0.f, 0.f, 0.f, 0.f, 0.f, 0.f};
  const float* xr = x + (size_t)tok * D_HID;
  for (int d = lane; d < D_HID; d += 32) {
    float xv = xr[d];
#pragma unroll
    for (int e = 0; e < 8; ++e) l[e] += xv * Wg[d * 8 + e];
  }
#pragma unroll
  for (int off = 16; off > 0; off >>= 1)
#pragma unroll
    for (int e = 0; e < 8; ++e) l[e] += __shfl_xor(l[e], off, 32);
  if (lane == 0) {
    float m = l[0];
#pragma unroll
    for (int e = 1; e < 8; ++e) m = fmaxf(m, l[e]);
    float p[8], s = 0.f;
#pragma unroll
    for (int e = 0; e < 8; ++e) { p[e] = __expf(l[e] - m); s += p[e]; }
    float inv = 1.f / s;
#pragma unroll
    for (int e = 0; e < 8; ++e) p[e] *= inv;
    int i0 = 0;
#pragma unroll
    for (int e = 1; e < 8; ++e) if (p[e] > p[i0]) i0 = e;
    int i1 = (i0 == 0) ? 1 : 0;
#pragma unroll
    for (int e = 0; e < 8; ++e) if (e != i0 && p[e] > p[i1]) i1 = e;
    int pos0 = atomicAdd(&cnt[i0], 1);
    slotList[i0 * N + pos0] = tok * 2;     wList[i0 * N + pos0] = p[i0];
    int pos1 = atomicAdd(&cnt[i1], 1);
    slotList[i1 * N + pos1] = tok * 2 + 1; wList[i1 * N + pos1] = p[i1];
  }
}

// ---------------- fused SwiGLU GEMM:  H = silu(A@Bg^T) * (A@Bu^T) * w ----------------
// A: [tokens][K] bf16 (row-gathered via slotList>>1 if routed)
// Bg/Bu: [e][Nout][K] bf16 (pre-transposed).  H: [slots][Nout] bf16.
// Block tile 128x64, 8 waves (4x2), wave tile 32x32, two accum sets.
// Double-buffered LDS; async global->LDS when available (batch k+32 in flight
// behind the WMMAs of batch k).
__global__ __launch_bounds__(256) void k_swiglu_gemm(
    const u16* __restrict__ A, const u16* __restrict__ Bg, const u16* __restrict__ Bu,
    u16* __restrict__ H, int K, int Nout, int Mdense,
    const int* __restrict__ cnt, const int* __restrict__ slotList,
    const float* __restrict__ wList) {
  const int e = blockIdx.z;
  const int M = cnt ? cnt[e] : Mdense;
  if ((int)(blockIdx.y * 128) >= M) return;
  const int lbase = e * Mdense;                // list pitch = Mdense (= N)
  const u16* BgE = Bg + (size_t)e * Nout * K;
  const u16* BuE = Bu + (size_t)e * Nout * K;

  const int ABUF = 128 * LPAD, BBUF = 64 * LPAD;
  __shared__ u16 As[2 * 128 * LPAD];
  __shared__ u16 Bgs[2 * 64 * LPAD];
  __shared__ u16 Bus[2 * 64 * LPAD];

  const int tid = threadIdx.x;
  const int lane = tid & 31, wid = tid >> 5;
  const int waveM = wid >> 1, waveN = wid & 1;
  const int half = lane >> 4, l16 = lane & 15;

  // A staging map: 2 x 16B chunks per thread (128 rows x 32 K, row = 4 chunks)
  size_t aSrc[2]; int aDst[2];
#pragma unroll
  for (int t = 0; t < 2; ++t) {
    int c = tid * 2 + t;
    int row = c >> 2, c16 = c & 3;
    int rg = blockIdx.y * 128 + row;
    int tok;
    if (slotList) tok = (rg < M) ? (slotList[lbase + rg] >> 1) : 0;
    else          tok = (rg < M) ? rg : 0;
    aSrc[t] = (size_t)tok * K + c16 * 8;
    aDst[t] = row * LPAD + c16 * 8;
  }
  // B staging map: 1 chunk per thread (64 rows x 32 K)
  int brow = tid >> 2, bc16 = tid & 3;
  size_t bSrc = (size_t)(blockIdx.x * 64 + brow) * K + bc16 * 8;
  int bDst = brow * LPAD + bc16 * 8;

  v8f accG[2][2] = {};
  v8f accU[2][2] = {};

  // prologue: batch k=0 into buffer 0
  stage16(&As[aDst[0]], &A[aSrc[0]]);
  stage16(&As[aDst[1]], &A[aSrc[1]]);
  stage16(&Bgs[bDst], &BgE[bSrc]);
  stage16(&Bus[bDst], &BuE[bSrc]);

  int buf = 0;
  for (int k0 = 0; k0 < K; k0 += KSTEP) {
    const int nxt = k0 + KSTEP;
    if (nxt < K) {                   // issue next batch into the other buffer
      int nb = buf ^ 1;
      stage16(&As[nb * ABUF + aDst[0]], &A[aSrc[0] + nxt]);
      stage16(&As[nb * ABUF + aDst[1]], &A[aSrc[1] + nxt]);
      stage16(&Bgs[nb * BBUF + bDst], &BgE[bSrc + nxt]);
      stage16(&Bus[nb * BBUF + bDst], &BuE[bSrc + nxt]);
      wait_stage<4>();               // current batch complete (only newest 4 pending)
    } else {
      wait_stage<0>();
    }
    __syncthreads();                 // all waves' chunks for this batch landed

    FragBF a[2], bg[2], bu[2];
#pragma unroll
    for (int mt = 0; mt < 2; ++mt) {
      const u16* p = &As[buf * ABUF + (waveM * 32 + mt * 16 + l16) * LPAD + half * 8];
      a[mt].h[0] = *(const v8bf*)p;
      a[mt].h[1] = *(const v8bf*)(p + 16);
    }
#pragma unroll
    for (int nt = 0; nt < 2; ++nt) {
      const u16* pg = &Bgs[buf * BBUF + (waveN * 32 + nt * 16 + l16) * LPAD + half * 8];
      bg[nt].h[0] = *(const v8bf*)pg;
      bg[nt].h[1] = *(const v8bf*)(pg + 16);
      const u16* pu = &Bus[buf * BBUF + (waveN * 32 + nt * 16 + l16) * LPAD + half * 8];
      bu[nt].h[0] = *(const v8bf*)pu;
      bu[nt].h[1] = *(const v8bf*)(pu + 16);
    }
#pragma unroll
    for (int mt = 0; mt < 2; ++mt)
#pragma unroll
      for (int nt = 0; nt < 2; ++nt) {
        accG[mt][nt] = wmma_bf16(a[mt].v, bg[nt].v, accG[mt][nt]);
        accU[mt][nt] = wmma_bf16(a[mt].v, bu[nt].v, accU[mt][nt]);
      }
    __syncthreads();                 // done reading buf before it is overwritten
    buf ^= 1;
  }

  const int rowBase = blockIdx.y * 128 + waveM * 32;
  const int colBase = blockIdx.x * 64 + waveN * 32;
#pragma unroll
  for (int mt = 0; mt < 2; ++mt) {
#pragma unroll
    for (int j = 0; j < 8; ++j) {
      int r = rowBase + mt * 16 + half * 8 + j;
      if (r >= M) continue;
      int srow = slotList ? slotList[lbase + r] : r;
      float w = wList ? wList[lbase + r] : 1.0f;
#pragma unroll
      for (int nt = 0; nt < 2; ++nt) {
        float g = accG[mt][nt][j];
        float u = accU[mt][nt][j];
        float h = (g / (1.0f + __expf(-g))) * u * w;
        H[(size_t)srow * Nout + colBase + nt * 16 + l16] = f2bf(h);
      }
    }
  }
}

// ---------------- down-projection GEMM: Out[row] = A[row] @ B^T, fp32 out ----------------
__global__ __launch_bounds__(256) void k_down_gemm(
    const u16* __restrict__ A, const u16* __restrict__ B,
    float* __restrict__ Out, int K, int Nout, int Mdense,
    const int* __restrict__ cnt, const int* __restrict__ slotList) {
  const int e = blockIdx.z;
  const int M = cnt ? cnt[e] : Mdense;
  if ((int)(blockIdx.y * 128) >= M) return;
  const int lbase = e * Mdense;
  const u16* BE = B + (size_t)e * Nout * K;

  const int ABUF = 128 * LPAD, BBUF = 64 * LPAD;
  __shared__ u16 As[2 * 128 * LPAD];
  __shared__ u16 Bs[2 * 64 * LPAD];

  const int tid = threadIdx.x;
  const int lane = tid & 31, wid = tid >> 5;
  const int waveM = wid >> 1, waveN = wid & 1;
  const int half = lane >> 4, l16 = lane & 15;

  size_t aSrc[2]; int aDst[2];
#pragma unroll
  for (int t = 0; t < 2; ++t) {
    int c = tid * 2 + t;
    int row = c >> 2, c16 = c & 3;
    int rg = blockIdx.y * 128 + row;
    int ridx;
    if (slotList) ridx = (rg < M) ? slotList[lbase + rg] : 0;   // gather H rows by slot
    else          ridx = (rg < M) ? rg : 0;
    aSrc[t] = (size_t)ridx * K + c16 * 8;
    aDst[t] = row * LPAD + c16 * 8;
  }
  int brow = tid >> 2, bc16 = tid & 3;
  size_t bSrc = (size_t)(blockIdx.x * 64 + brow) * K + bc16 * 8;
  int bDst = brow * LPAD + bc16 * 8;

  v8f acc[2][2] = {};

  stage16(&As[aDst[0]], &A[aSrc[0]]);
  stage16(&As[aDst[1]], &A[aSrc[1]]);
  stage16(&Bs[bDst], &BE[bSrc]);

  int buf = 0;
  for (int k0 = 0; k0 < K; k0 += KSTEP) {
    const int nxt = k0 + KSTEP;
    if (nxt < K) {
      int nb = buf ^ 1;
      stage16(&As[nb * ABUF + aDst[0]], &A[aSrc[0] + nxt]);
      stage16(&As[nb * ABUF + aDst[1]], &A[aSrc[1] + nxt]);
      stage16(&Bs[nb * BBUF + bDst], &BE[bSrc + nxt]);
      wait_stage<3>();
    } else {
      wait_stage<0>();
    }
    __syncthreads();

    FragBF a[2], b[2];
#pragma unroll
    for (int mt = 0; mt < 2; ++mt) {
      const u16* p = &As[buf * ABUF + (waveM * 32 + mt * 16 + l16) * LPAD + half * 8];
      a[mt].h[0] = *(const v8bf*)p;
      a[mt].h[1] = *(const v8bf*)(p + 16);
    }
#pragma unroll
    for (int nt = 0; nt < 2; ++nt) {
      const u16* p = &Bs[buf * BBUF + (waveN * 32 + nt * 16 + l16) * LPAD + half * 8];
      b[nt].h[0] = *(const v8bf*)p;
      b[nt].h[1] = *(const v8bf*)(p + 16);
    }
#pragma unroll
    for (int mt = 0; mt < 2; ++mt)
#pragma unroll
      for (int nt = 0; nt < 2; ++nt)
        acc[mt][nt] = wmma_bf16(a[mt].v, b[nt].v, acc[mt][nt]);
    __syncthreads();
    buf ^= 1;
  }

  const int rowBase = blockIdx.y * 128 + waveM * 32;
  const int colBase = blockIdx.x * 64 + waveN * 32;
#pragma unroll
  for (int mt = 0; mt < 2; ++mt) {
#pragma unroll
    for (int j = 0; j < 8; ++j) {
      int r = rowBase + mt * 16 + half * 8 + j;
      if (r >= M) continue;
      int srow = slotList ? slotList[lbase + r] : r;   // scatter by slot
#pragma unroll
      for (int nt = 0; nt < 2; ++nt)
        Out[(size_t)srow * Nout + colBase + nt * 16 + l16] = acc[mt][nt][j];
    }
  }
}

// ---------------- final combine: out[n] = shared[n] + slot(2n) + slot(2n+1) ----------------
__global__ void k_combine(float* __restrict__ out, const float* __restrict__ outR, int ND) {
  int i = blockIdx.x * blockDim.x + threadIdx.x;
  if (i >= ND) return;
  int n = i / D_HID, d = i - n * D_HID;
  out[i] += outR[(size_t)(2 * n) * D_HID + d] + outR[(size_t)(2 * n + 1) * D_HID + d];
}

// ---------------- host launch ----------------

extern "C" void kernel_launch(void* const* d_in, const int* in_sizes, int n_in,
                              void* d_out, int out_size, void* d_ws, size_t ws_size,
                              hipStream_t stream) {
  (void)n_in; (void)out_size; (void)ws_size;
  const float* x    = (const float*)d_in[0];
  const float* W_g  = (const float*)d_in[1];
  const float* Wg_e = (const float*)d_in[2];
  const float* Wu_e = (const float*)d_in[3];
  const float* Wd_e = (const float*)d_in[4];
  const float* Wg_s = (const float*)d_in[5];
  const float* Wu_s = (const float*)d_in[6];
  const float* Wd_s = (const float*)d_in[7];

  const int N = in_sizes[0] / D_HID;   // 2048 tokens

  char* ws = (char*)d_ws;
  size_t off = 0;
  auto carve = [&](size_t bytes) -> char* {
    char* p = ws + off;
    off = (off + bytes + 255) & ~(size_t)255;
    return p;
  };
  u16*   xbf  = (u16*)carve((size_t)N * D_HID * 2);
  u16*   WgsT = (u16*)carve((size_t)D_SH * D_HID * 2);
  u16*   WusT = (u16*)carve((size_t)D_SH * D_HID * 2);
  u16*   WdsT = (u16*)carve((size_t)D_HID * D_SH * 2);
  u16*   WgeT = (u16*)carve((size_t)N_EXP * D_EXP * D_HID * 2);
  u16*   WueT = (u16*)carve((size_t)N_EXP * D_EXP * D_HID * 2);
  u16*   WdeT = (u16*)carve((size_t)N_EXP * D_HID * D_EXP * 2);
  u16*   Hs   = (u16*)carve((size_t)N * D_SH * 2);
  u16*   Hr   = (u16*)carve((size_t)N * 2 * D_EXP * 2);
  float* outR = (float*)carve((size_t)N * 2 * D_HID * 4);
  int*   cnt  = (int*)carve(64);
  int*   slotList = (int*)carve((size_t)N_EXP * N * 4);
  float* wList    = (float*)carve((size_t)N_EXP * N * 4);

  float* out = (float*)d_out;

  k_zero<<<1, 32, 0, stream>>>(cnt, 8);
  k_f32_to_bf16<<<(N * D_HID + 255) / 256, 256, 0, stream>>>(x, xbf, N * D_HID);
  // weights -> bf16, transposed to [Nout][K]
  k_transpose_bf16<<<dim3(D_SH / 32, D_HID / 32, 1), 256, 0, stream>>>(Wg_s, WgsT, D_HID, D_SH);
  k_transpose_bf16<<<dim3(D_SH / 32, D_HID / 32, 1), 256, 0, stream>>>(Wu_s, WusT, D_HID, D_SH);
  k_transpose_bf16<<<dim3(D_HID / 32, D_SH / 32, 1), 256, 0, stream>>>(Wd_s, WdsT, D_SH, D_HID);
  k_transpose_bf16<<<dim3(D_EXP / 32, D_HID / 32, N_EXP), 256, 0, stream>>>(Wg_e, WgeT, D_HID, D_EXP);
  k_transpose_bf16<<<dim3(D_EXP / 32, D_HID / 32, N_EXP), 256, 0, stream>>>(Wu_e, WueT, D_HID, D_EXP);
  k_transpose_bf16<<<dim3(D_HID / 32, D_EXP / 32, N_EXP), 256, 0, stream>>>(Wd_e, WdeT, D_EXP, D_HID);

  k_gate<<<(N + 7) / 8, 256, 0, stream>>>(x, W_g, cnt, slotList, wList, N);

  // shared expert: SwiGLU then down-proj (writes d_out fully)
  k_swiglu_gemm<<<dim3(D_SH / 64, N / 128, 1), 256, 0, stream>>>(
      xbf, WgsT, WusT, Hs, D_HID, D_SH, N, nullptr, nullptr, nullptr);
  k_down_gemm<<<dim3(D_HID / 64, N / 128, 1), 256, 0, stream>>>(
      Hs, WdsT, out, D_SH, D_HID, N, nullptr, nullptr);

  // routed experts: compacted per-expert GEMMs (early-exit past cnt[e])
  k_swiglu_gemm<<<dim3(D_EXP / 64, N / 128, N_EXP), 256, 0, stream>>>(
      xbf, WgeT, WueT, Hr, D_HID, D_EXP, N, cnt, slotList, wList);
  k_down_gemm<<<dim3(D_HID / 64, N / 128, N_EXP), 256, 0, stream>>>(
      Hr, WdeT, outR, D_EXP, D_HID, N, cnt, slotList);

  k_combine<<<(N * D_HID + 255) / 256, 256, 0, stream>>>(out, outR, N * D_HID);
}
